// MultiHeadAttention_70806830841940
// MI455X (gfx1250) — matrix operand order repaired and checked by reading, not compiled, not run on hardware
//
#include <hip/hip_runtime.h>
#include <hip/hip_bf16.h>

// ---------------------------------------------------------------------------
// MI455X (gfx1250) fused multi-head attention, bf16 WMMA path, wide LDS I/O.
//   D_MODEL=1024, H=16, depth=64, B=2, S=2048, M = B*S = 4096
// ---------------------------------------------------------------------------

typedef __attribute__((ext_vector_type(16))) __bf16 v16bf;
typedef __attribute__((ext_vector_type(8)))  __bf16 v8bf;
typedef __attribute__((ext_vector_type(4)))  __bf16 v4bf;
typedef __attribute__((ext_vector_type(8)))  float  v8f;

static constexpr int kD  = 1024;   // D_MODEL
static constexpr int kS  = 2048;   // sequence length
static constexpr int kH  = 16;     // heads
static constexpr int kDp = 64;     // depth per head
static constexpr int kM  = 4096;   // B*S rows

__device__ __forceinline__ unsigned short f2bits(float f) {
    union { float f; unsigned u; } x; x.f = f;
    unsigned u = x.u + 0x7FFFu + ((x.u >> 16) & 1u);   // round-to-nearest-even
    return (unsigned short)(u >> 16);
}
__device__ __forceinline__ __bf16 bits2bf(unsigned short s) {
    __bf16 b; __builtin_memcpy(&b, &s, 2); return b;
}
__device__ __forceinline__ __bf16 f2bf(float f) { return bits2bf(f2bits(f)); }

__device__ __forceinline__ v16bf cat8(v8bf lo, v8bf hi) {
    return __builtin_shufflevector(lo, hi, 0,1,2,3,4,5,6,7,8,9,10,11,12,13,14,15);
}
__device__ __forceinline__ v8f vzero8() {
    return (v8f){0.f,0.f,0.f,0.f,0.f,0.f,0.f,0.f};
}

// ---------------------------------------------------------------------------
// Pre-pass: f32 -> bf16 elementwise (activations).
// ---------------------------------------------------------------------------
__global__ __launch_bounds__(256) void cvt_bf16_kernel(
    const float* __restrict__ src, __bf16* __restrict__ dst)
{
    int i = (blockIdx.x * 256 + threadIdx.x) * 4;
    float4 v = *(const float4*)(src + i);
    v4bf o = { f2bf(v.x), f2bf(v.y), f2bf(v.z), f2bf(v.w) };
    *(v4bf*)(dst + i) = o;
}

// ---------------------------------------------------------------------------
// Pre-pass: W (K x N, f32) -> Wt (N x K, bf16), 32x32 LDS tile transpose.
// grid (K/32, N/32), block (32, 8)
// ---------------------------------------------------------------------------
__global__ __launch_bounds__(256) void wT_kernel(
    const float* __restrict__ W, __bf16* __restrict__ Wt)
{
    __shared__ __bf16 tile[32][33];
    const int tx = threadIdx.x, ty = threadIdx.y;
    const int kB = blockIdx.x * 32, nB = blockIdx.y * 32;
#pragma unroll
    for (int i = 0; i < 32; i += 8)
        tile[ty + i][tx] = f2bf(W[(kB + ty + i) * kD + nB + tx]);
    __syncthreads();
#pragma unroll
    for (int i = 0; i < 32; i += 8)
        Wt[(nB + ty + i) * kD + kB + tx] = tile[tx][ty + i];
}

// ---------------------------------------------------------------------------
// GEMM + bias: C(4096x1024) = A(bf16, MxK) * Wt(bf16, NxK)^T + bias
// MODE 0: out f32 row-major (final projection -> d_out)
// MODE 1: out bf16 head layout   ((b*16+h)*2048+s)*64+d      (Q, K)
// MODE 2: out bf16 head-T layout ((b*16+h)*64+d)*2048+s      (V)
// Block: 128 threads = 4 waves; block tile 128(M) x 64(N); K-step 32.
// Wave tile 32(M) x 64(N): 8 WMMAs per K-step, B fragments reused 2x.
// ---------------------------------------------------------------------------
template <int MODE>
__global__ __launch_bounds__(128) void gemm_bias_kernel(
    const __bf16* __restrict__ A, const __bf16* __restrict__ Wt,
    const float* __restrict__ bias, void* __restrict__ outp)
{
    __shared__ alignas(16) __bf16 sA[128][40];  // 128 rows x 32 k, 80B stride
    __shared__ alignas(16) __bf16 sB[64][40];   // 64 n     x 32 k

    const int tid   = threadIdx.x;
    const int wave  = tid >> 5;
    const int lane  = tid & 31;
    const int n16   = lane & 15;
    const int hf    = lane >> 4;
    const int mBase = blockIdx.x * 128;
    const int nBase = blockIdx.y * 64;

    // per-thread staging offsets (base pointers stay kernel args -> global AS)
    int offA[4];
    __bf16* la[4];
#pragma unroll
    for (int i = 0; i < 4; ++i) {
        int p = tid + i * 128;                 // 0..511 -> A tile chunks
        int r = p >> 2, seg = (p & 3) * 8;
        offA[i] = (mBase + r) * kD + seg;
        la[i]   = &sA[r][seg];
    }
    int offB[2];
    __bf16* lb[2];
#pragma unroll
    for (int i = 0; i < 2; ++i) {
        int p = tid + i * 128;                 // 0..255 -> B tile chunks
        int r = p >> 2, seg = (p & 3) * 8;
        offB[i] = (nBase + r) * kD + seg;
        lb[i]   = &sB[r][seg];
    }

    v8f acc[2][4];
#pragma unroll
    for (int g = 0; g < 2; ++g)
#pragma unroll
        for (int t = 0; t < 4; ++t) acc[g][t] = vzero8();

    for (int kb = 0; kb < kD; kb += 32) {
        __syncthreads();
        // issue ALL staging loads into distinct temps, then drain to LDS:
        // keeps loads pipelined (staggered s_wait_loadcnt) instead of
        // serializing each load->store round trip.
        v8bf ta0 = *(const v8bf*)(A  + offA[0] + kb);
        v8bf ta1 = *(const v8bf*)(A  + offA[1] + kb);
        v8bf ta2 = *(const v8bf*)(A  + offA[2] + kb);
        v8bf ta3 = *(const v8bf*)(A  + offA[3] + kb);
        v8bf tb0 = *(const v8bf*)(Wt + offB[0] + kb);
        v8bf tb1 = *(const v8bf*)(Wt + offB[1] + kb);
        *(v8bf*)la[0] = ta0;
        *(v8bf*)la[1] = ta1;
        *(v8bf*)la[2] = ta2;
        *(v8bf*)la[3] = ta3;
        *(v8bf*)lb[0] = tb0;
        *(v8bf*)lb[1] = tb1;
        if (kb + 32 < kD) {   // prefetch next K-slab (global_prefetch_b8)
            __builtin_prefetch(A  + offA[0] + kb + 32, 0, 1);
            __builtin_prefetch(Wt + offB[0] + kb + 32, 0, 1);
        }
        __syncthreads();

        // A fragments: lane = row m; e<8 -> k=8*hf+e ; e>=8 -> k=16+8*hf+e
        v16bf a[2];
#pragma unroll
        for (int g = 0; g < 2; ++g)
            a[g] = cat8(*(const v8bf*)&sA[wave * 32 + g * 16 + n16][8 * hf],
                        *(const v8bf*)&sA[wave * 32 + g * 16 + n16][16 + 8 * hf]);
#pragma unroll
        for (int t = 0; t < 4; ++t) {
            // B fragment: lane = col n; e<8 -> k=8*hf+e ; e>=8 -> k=16+8*hf+e
            v16bf b = cat8(*(const v8bf*)&sB[t * 16 + n16][8 * hf],
                           *(const v8bf*)&sB[t * 16 + n16][16 + 8 * hf]);
#pragma unroll
            for (int g = 0; g < 2; ++g)
                acc[g][t] = __builtin_amdgcn_wmma_f32_16x16x32_bf16(
                    false, a[g], false, b, (short)0, acc[g][t], false, false);
        }
    }

    // epilogue: C elem r -> row r+8*hf, col = lane&15   (branch-free per MODE)
#pragma unroll
    for (int t = 0; t < 4; ++t) {
        int col = nBase + t * 16 + n16;
        float bval = bias[col];
#pragma unroll
        for (int g = 0; g < 2; ++g)
#pragma unroll
            for (int r = 0; r < 8; ++r) {
                int Mrow = mBase + wave * 32 + g * 16 + r + 8 * hf;
                float v = acc[g][t][r] + bval;
                if constexpr (MODE == 0) {
                    ((float*)outp)[(size_t)Mrow * kD + col] = v;
                } else {
                    int b = Mrow >> 11, s = Mrow & (kS - 1);
                    int hh = col >> 6, d = col & (kDp - 1);
                    if constexpr (MODE == 1)
                        ((__bf16*)outp)[(((b * kH + hh) * kS) + s) * kDp + d] = f2bf(v);
                    else
                        ((__bf16*)outp)[(((b * kH + hh) * kDp) + d) * kS + s] = f2bf(v);
                }
            }
    }
}

// ---------------------------------------------------------------------------
// Flash attention: one 16-row Q tile per wave; 4 waves / block; 64 blocks.
// Qh/Kh bf16 (B,H,S,64); Vt bf16 (B,H,64,S); output Ao bf16 (B,S,1024).
// Key chunk = 64: 8 WMMAs (logits) + one softmax update + 8 WMMAs (P*V).
// ---------------------------------------------------------------------------
__global__ __launch_bounds__(128) void attn_kernel(
    const __bf16* __restrict__ Qh, const __bf16* __restrict__ Kh,
    const __bf16* __restrict__ Vt, const int* __restrict__ mask,
    __bf16* __restrict__ Ao)
{
    __shared__ alignas(16) __bf16 sP[4][16][64];   // per-wave P tile (16x64)

    const int tid  = threadIdx.x;
    const int wave = tid >> 5;
    const int lane = tid & 31;
    const int n16  = lane & 15;
    const int hf   = lane >> 4;

    const int tile  = blockIdx.x * 4 + wave;    // 0..255 q-tiles
    const int bh    = tile >> 7;                // b*16+h
    const int qt    = tile & 127;
    const int b     = bh >> 4;
    const int hh    = bh & 15;
    const int qBase = qt * 16;

    // Q fragments, depth chunks 0 and 32 (A layout; two 16B loads each)
    const __bf16* qrow = &Qh[(size_t)(bh * kS + qBase + n16) * kDp];
    v16bf qf[2];
#pragma unroll
    for (int c = 0; c < 2; ++c)
        qf[c] = cat8(*(const v8bf*)&qrow[c * 32 + 8 * hf],
                     *(const v8bf*)&qrow[c * 32 + 16 + 8 * hf]);

    // integer base offsets for K, V, mask streams (keeps global_load path)
    int kOff[4], vOff[4];
#pragma unroll
    for (int t = 0; t < 4; ++t) {
        kOff[t] = (bh * kS + t * 16 + n16) * kDp;   // + kc*64 per chunk
        vOff[t] = (bh * kDp + t * 16 + n16) * kS;   // + kc    per chunk
    }
    const int mOff = b * kS + n16;

    float mrun[8], lrun[8];
#pragma unroll
    for (int r = 0; r < 8; ++r) { mrun[r] = -3.0e38f; lrun[r] = 0.f; }
    v8f acc[4];
#pragma unroll
    for (int t = 0; t < 4; ++t) acc[t] = vzero8();

    for (int kc = 0; kc < kS; kc += 64) {
        const int kcd = kc * kDp;
        // ---- logits: four 16-key tiles, contracted over depth 64 ----
        v8f st[4];
#pragma unroll
        for (int t = 0; t < 4; ++t) {
            const __bf16* kr = Kh + kOff[t] + kcd;
            v16bf kb0 = cat8(*(const v8bf*)&kr[8 * hf],
                             *(const v8bf*)&kr[16 + 8 * hf]);
            v16bf kb1 = cat8(*(const v8bf*)&kr[32 + 8 * hf],
                             *(const v8bf*)&kr[48 + 8 * hf]);
            v8f c0 = vzero8();
            c0 = __builtin_amdgcn_wmma_f32_16x16x32_bf16(
                false, qf[0], false, kb0, (short)0, c0, false, false);
            c0 = __builtin_amdgcn_wmma_f32_16x16x32_bf16(
                false, qf[1], false, kb1, (short)0, c0, false, false);
            float mb = -1.0e9f * (float)mask[mOff + kc + t * 16];
#pragma unroll
            for (int r = 0; r < 8; ++r)
                c0[r] = fmaf(c0[r], 0.125f, mb);   // scale 1/sqrt(64) + mask
            st[t] = c0;
        }

        // ---- online softmax update (rows r+8*hf; reduce over 16 lanes) ----
#pragma unroll
        for (int r = 0; r < 8; ++r) {
            float vmax = fmaxf(fmaxf(st[0][r], st[1][r]),
                               fmaxf(st[2][r], st[3][r]));
#pragma unroll
            for (int m = 1; m < 16; m <<= 1)
                vmax = fmaxf(vmax, __shfl_xor(vmax, m, 32));
            float mnew  = fmaxf(mrun[r], vmax);
            float alpha = __expf(mrun[r] - mnew);
            float p0 = __expf(st[0][r] - mnew);
            float p1 = __expf(st[1][r] - mnew);
            float p2 = __expf(st[2][r] - mnew);
            float p3 = __expf(st[3][r] - mnew);
            float rs = (p0 + p1) + (p2 + p3);
#pragma unroll
            for (int m = 1; m < 16; m <<= 1)
                rs += __shfl_xor(rs, m, 32);
            lrun[r] = lrun[r] * alpha + rs;
            mrun[r] = mnew;
#pragma unroll
            for (int t = 0; t < 4; ++t) acc[t][r] *= alpha;
            int row = r + 8 * hf;                  // transpose through LDS
            sP[wave][row][n16]      = f2bf(p0);
            sP[wave][row][16 + n16] = f2bf(p1);
            sP[wave][row][32 + n16] = f2bf(p2);
            sP[wave][row][48 + n16] = f2bf(p3);
        }
        __syncthreads();   // uniform trip count across all 4 waves

        // ---- P fragments (A layout, key chunks 0..31 and 32..63) ----
        v16bf pa0 = cat8(*(const v8bf*)&sP[wave][n16][8 * hf],
                         *(const v8bf*)&sP[wave][n16][16 + 8 * hf]);
        v16bf pa1 = cat8(*(const v8bf*)&sP[wave][n16][32 + 8 * hf],
                         *(const v8bf*)&sP[wave][n16][48 + 8 * hf]);
        // ---- O += P * V over 4 depth tiles (V transposed: contiguous) ----
#pragma unroll
        for (int t = 0; t < 4; ++t) {
            const __bf16* vr = Vt + vOff[t] + kc;
            v16bf vb0 = cat8(*(const v8bf*)&vr[8 * hf],
                             *(const v8bf*)&vr[16 + 8 * hf]);
            v16bf vb1 = cat8(*(const v8bf*)&vr[32 + 8 * hf],
                             *(const v8bf*)&vr[48 + 8 * hf]);
            acc[t] = __builtin_amdgcn_wmma_f32_16x16x32_bf16(
                false, pa0, false, vb0, (short)0, acc[t], false, false);
            acc[t] = __builtin_amdgcn_wmma_f32_16x16x32_bf16(
                false, pa1, false, vb1, (short)0, acc[t], false, false);
        }
    }

    // ---- normalize + store bf16 into (B,S,D_MODEL) row-major ----
#pragma unroll
    for (int r = 0; r < 8; ++r) {
        float inv = 1.0f / lrun[r];
        int q = qBase + r + 8 * hf;
#pragma unroll
        for (int t = 0; t < 4; ++t)
            Ao[(size_t)(b * kS + q) * kD + hh * kDp + t * 16 + n16] =
                f2bf(acc[t][r] * inv);
    }
}

// ---------------------------------------------------------------------------
// Launch
// ---------------------------------------------------------------------------
extern "C" void kernel_launch(void* const* d_in, const int* in_sizes, int n_in,
                              void* d_out, int out_size, void* d_ws, size_t ws_size,
                              hipStream_t stream) {
    const float* q    = (const float*)d_in[0];
    const float* k    = (const float*)d_in[1];
    const float* v    = (const float*)d_in[2];
    const int*   mask = (const int*)  d_in[3];
    const float* wq   = (const float*)d_in[4];
    const float* bq   = (const float*)d_in[5];
    const float* wk   = (const float*)d_in[6];
    const float* bk   = (const float*)d_in[7];
    const float* wvp  = (const float*)d_in[8];
    const float* bv   = (const float*)d_in[9];
    const float* wo   = (const float*)d_in[10];
    const float* bo   = (const float*)d_in[11];

    const size_t act = (size_t)kM * kD;     // 4,194,304 elems
    const size_t wsz = (size_t)kD * kD;     // 1,048,576 elems
    __bf16* p   = (__bf16*)d_ws;
    __bf16* qb  = p;                 p += act;
    __bf16* kb_ = p;                 p += act;
    __bf16* vb  = p;                 p += act;
    __bf16* wqT = p;                 p += wsz;
    __bf16* wkT = p;                 p += wsz;
    __bf16* wvT = p;                 p += wsz;
    __bf16* woT = p;                 p += wsz;
    __bf16* Qh  = p;                 p += act;
    __bf16* Kh  = p;                 p += act;
    __bf16* Vt  = p;                 p += act;
    __bf16* Ao  = p;                 p += act;

    // pre-pass: activations -> bf16 ; weights -> bf16 transposed
    dim3 cvtGrid((unsigned)(act / 4 / 256));
    cvt_bf16_kernel<<<cvtGrid, 256, 0, stream>>>(q, qb);
    cvt_bf16_kernel<<<cvtGrid, 256, 0, stream>>>(k, kb_);
    cvt_bf16_kernel<<<cvtGrid, 256, 0, stream>>>(v, vb);
    dim3 tGrid(kD / 32, kD / 32), tBlk(32, 8);
    wT_kernel<<<tGrid, tBlk, 0, stream>>>(wq,  wqT);
    wT_kernel<<<tGrid, tBlk, 0, stream>>>(wk,  wkT);
    wT_kernel<<<tGrid, tBlk, 0, stream>>>(wvp, wvT);
    wT_kernel<<<tGrid, tBlk, 0, stream>>>(wo,  woT);

    dim3 gblk(128);
    dim3 ggrid(kM / 128, kD / 64);  // 32 x 16
    gemm_bias_kernel<1><<<ggrid, gblk, 0, stream>>>(qb,  wqT, bq, Qh);
    gemm_bias_kernel<1><<<ggrid, gblk, 0, stream>>>(kb_, wkT, bk, Kh);
    gemm_bias_kernel<2><<<ggrid, gblk, 0, stream>>>(vb,  wvT, bv, Vt);

    attn_kernel<<<dim3(kM / 16 / 4), gblk, 0, stream>>>(Qh, Kh, Vt, mask, Ao);

    gemm_bias_kernel<0><<<ggrid, gblk, 0, stream>>>(Ao, woT, bo, d_out);
}
